// Attention_4810363372688
// MI455X (gfx1250) — matrix operand order repaired and checked by reading, not compiled, not run on hardware
//
#include <hip/hip_runtime.h>
#include <hip/hip_bf16.h>

typedef __bf16 bf16_t;
typedef __attribute__((ext_vector_type(16))) __bf16 bf16x16;
typedef __attribute__((ext_vector_type(8)))  __bf16 bf16x8;
typedef __attribute__((ext_vector_type(8)))  float  floatx8;
typedef __attribute__((ext_vector_type(4)))  float  floatx4;

// ---------------------------------------------------------------------------
// WMMA helper: D = A(16x32 bf16) * B(32x16 bf16) + C(16x16 f32)
// ---------------------------------------------------------------------------
__device__ __forceinline__ floatx8 wmma_bf16(bf16x16 a, bf16x16 b, floatx8 c) {
  return __builtin_amdgcn_wmma_f32_16x16x32_bf16(
      /*neg_a=*/false, a, /*neg_b=*/false, b,
      /*c_mod=*/(short)0, c, /*reuse_a=*/false, /*reuse_b=*/false);
}

// A fragment (16x32, M x K) from row-major bf16 memory.
// lane l: row = l&15 ; K = {kb..kb+7, kb+16..kb+23}, kb = 8*(l>>4)
__device__ __forceinline__ bf16x16 ldA(const bf16_t* base, int ld, int lane) {
  const bf16_t* p = base + (size_t)(lane & 15) * ld + ((lane >> 4) << 3);
  bf16x8 lo = *(const bf16x8*)p;
  bf16x8 hi = *(const bf16x8*)(p + 16);
  bf16x16 f;
#pragma unroll
  for (int i = 0; i < 8; ++i) { f[i] = lo[i]; f[8 + i] = hi[i]; }
  return f;
}

// A fragment converted on the fly from fp32 memory.
__device__ __forceinline__ bf16x16 ldA_f32(const float* base, int ld, int lane) {
  const float* p = base + (size_t)(lane & 15) * ld + ((lane >> 4) << 3);
  floatx4 a0 = *(const floatx4*)p;
  floatx4 a1 = *(const floatx4*)(p + 4);
  floatx4 b0 = *(const floatx4*)(p + 16);
  floatx4 b1 = *(const floatx4*)(p + 20);
  bf16x16 f;
#pragma unroll
  for (int i = 0; i < 4; ++i) {
    f[i]      = (bf16_t)a0[i];
    f[4 + i]  = (bf16_t)a1[i];
    f[8 + i]  = (bf16_t)b0[i];
    f[12 + i] = (bf16_t)b1[i];
  }
  return f;
}

// B fragment (32x16, K x N) from memory holding B^T row-major (N x K).
// lane l: col n = l&15 ; K = {kb..kb+7, kb+16..kb+23}, kb = 8*(l>>4)
__device__ __forceinline__ bf16x16 ldB(const bf16_t* base, int ld, int lane) {
  const bf16_t* p = base + (size_t)(lane & 15) * ld + ((lane >> 4) << 3);
  bf16x8 lo = *(const bf16x8*)p;
  bf16x8 hi = *(const bf16x8*)(p + 16);
  bf16x16 f;
#pragma unroll
  for (int i = 0; i < 8; ++i) { f[i] = lo[i]; f[8 + i] = hi[i]; }
  return f;
}

// ---------------------------------------------------------------------------
// Prep kernels (weight transpose + f32->bf16)
// ---------------------------------------------------------------------------
__global__ void k_conv_wT(const float* __restrict__ W, bf16_t* __restrict__ WT) {
  int o = blockIdx.x * 256 + threadIdx.x;          // o = hd*1024 + k
  if (o >= 1024 * 1024) return;
  int k = o & 1023, hd = o >> 10;
  WT[o] = (bf16_t)W[(size_t)k * 1024 + hd];
}
__global__ void k_conv_woT(const float* __restrict__ Wo, bf16_t* __restrict__ WoT) {
  int o = blockIdx.x * 256 + threadIdx.x;          // o = n*1024 + hd
  if (o >= 1024 * 1024) return;
  int hd = o & 1023, n = o >> 10;
  WoT[o] = (bf16_t)Wo[(size_t)hd * 1024 + n];
}
__global__ void k_conv_proj(const float* __restrict__ p, bf16_t* __restrict__ pb) {
  int o = blockIdx.x * 256 + threadIdx.x;
  if (o < 128 * 64) pb[o] = (bf16_t)(p[o] * 0.08838834764831845f);  // 1/sqrt(128)
}
__global__ void k_zero_bf16(bf16_t* __restrict__ p, int n) {
  int o = blockIdx.x * 256 + threadIdx.x;
  if (o < n) p[o] = (bf16_t)0.0f;
}

// ---------------------------------------------------------------------------
// Fused QKV projection + ReLU feature map. 32 rows per wave,
// double-buffered / unroll-2 pipelined main loop.
// grid: (B*L/32, H, 3)  z=0: qp, z=1: kp (transposed), z=2: v (transposed)
// ---------------------------------------------------------------------------
__global__ __launch_bounds__(32) void k_qkv_feat(
    const float* __restrict__ Xq, const float* __restrict__ Xs,
    const bf16_t* __restrict__ WqT, const bf16_t* __restrict__ WkT,
    const bf16_t* __restrict__ WvT, const bf16_t* __restrict__ projb,
    bf16_t* __restrict__ qp, bf16_t* __restrict__ kpT, bf16_t* __restrict__ vT) {
  __shared__ bf16_t smemT[16 * 64];    // stage-1 tile, C-layout -> A-layout
  __shared__ bf16_t smemO[16 * 128];   // qp store staging (coalesced)
  const int lane = threadIdx.x;
  const int t = blockIdx.x;            // 32-row tile  (B*L/32 = 512)
  const int h = blockIdx.y;
  const int z = blockIdx.z;
  const int b = t >> 7;                // 128 tiles per batch
  const int l0 = (t & 127) * 32;

  const float* X = (z == 0) ? Xq : Xs;
  const bf16_t* WT = (z == 0) ? WqT : (z == 1 ? WkT : WvT);
  const float* Xt = X + ((size_t)(b * 4096 + l0)) * 1024;
  const bf16_t* Wh = WT + (size_t)h * 64 * 1024;

  // ---- stage 1: T[32x64] = X[32x1024] @ W_h ----
  floatx8 acc[8] = {};
  bf16x16 Af[2][2], Bf[4][2];
#define LOAD_S1(p, kk)                                            \
  do {                                                            \
    Af[0][p] = ldA_f32(Xt + (kk), 1024, lane);                    \
    Af[1][p] = ldA_f32(Xt + 16 * 1024 + (kk), 1024, lane);        \
    Bf[0][p] = ldB(Wh + 0 * 16 * 1024 + (kk), 1024, lane);        \
    Bf[1][p] = ldB(Wh + 1 * 16 * 1024 + (kk), 1024, lane);        \
    Bf[2][p] = ldB(Wh + 2 * 16 * 1024 + (kk), 1024, lane);        \
    Bf[3][p] = ldB(Wh + 3 * 16 * 1024 + (kk), 1024, lane);        \
  } while (0)
#define COMP_S1(p)                                                \
  do {                                                            \
    acc[0] = wmma_bf16(Af[0][p], Bf[0][p], acc[0]);               \
    acc[1] = wmma_bf16(Af[0][p], Bf[1][p], acc[1]);               \
    acc[2] = wmma_bf16(Af[0][p], Bf[2][p], acc[2]);               \
    acc[3] = wmma_bf16(Af[0][p], Bf[3][p], acc[3]);               \
    acc[4] = wmma_bf16(Af[1][p], Bf[0][p], acc[4]);               \
    acc[5] = wmma_bf16(Af[1][p], Bf[1][p], acc[5]);               \
    acc[6] = wmma_bf16(Af[1][p], Bf[2][p], acc[6]);               \
    acc[7] = wmma_bf16(Af[1][p], Bf[3][p], acc[7]);               \
  } while (0)
  LOAD_S1(0, 0);
  for (int kk = 0; kk < 1024 - 64; kk += 64) {
    LOAD_S1(1, kk + 32);
    COMP_S1(0);
    LOAD_S1(0, kk + 64);
    COMP_S1(1);
  }
  LOAD_S1(1, 1024 - 32);
  COMP_S1(0);
  COMP_S1(1);
#undef LOAD_S1
#undef COMP_S1

  const int bh = b * 16 + h;
  const int cn = lane & 15;
  const int rb = (lane >> 4) << 3;

  if (z == 2) {  // store V transposed: vT[bh][d][l]
#pragma unroll
    for (int hlf = 0; hlf < 2; ++hlf)
#pragma unroll
      for (int n = 0; n < 4; ++n) {
        int d = n * 16 + cn;
        bf16x8 o;
#pragma unroll
        for (int r = 0; r < 8; ++r) o[r] = (bf16_t)acc[hlf * 4 + n][r];
        *(bf16x8*)(vT + ((size_t)bh * 64 + d) * 4096 + l0 + hlf * 16 + rb) = o;
      }
    return;
  }

  // ---- stage 2 per 16-row half: F = relu(T @ proj^T) + eps ----
#pragma unroll 1
  for (int hlf = 0; hlf < 2; ++hlf) {
    asm volatile("s_wait_dscnt 0" ::: "memory");  // smem reuse hazard fence
#pragma unroll
    for (int n = 0; n < 4; ++n) {
      int col = n * 16 + cn;
#pragma unroll
      for (int r = 0; r < 8; ++r)
        smemT[(rb + r) * 64 + col] = (bf16_t)acc[hlf * 4 + n][r];
    }
    asm volatile("s_wait_dscnt 0" ::: "memory");

    floatx8 f[8] = {};
    bf16x16 ta = ldA(smemT, 64, lane);
#pragma unroll
    for (int n = 0; n < 8; ++n)
      f[n] = wmma_bf16(ta, ldB(projb + n * 16 * 64, 64, lane), f[n]);
    ta = ldA(smemT + 32, 64, lane);
#pragma unroll
    for (int n = 0; n < 8; ++n)
      f[n] = wmma_bf16(ta, ldB(projb + n * 16 * 64 + 32, 64, lane), f[n]);

    if (z == 0) {
      // stage through LDS -> fully contiguous 16x128 bf16 block store
#pragma unroll
      for (int n = 0; n < 8; ++n) {
        int m = n * 16 + cn;
#pragma unroll
        for (int r = 0; r < 8; ++r)
          smemO[(rb + r) * 128 + m] = (bf16_t)(fmaxf(f[n][r], 0.0f) + 1e-3f);
      }
      asm volatile("s_wait_dscnt 0" ::: "memory");
      bf16_t* dst = qp + ((size_t)bh * 4096 + l0 + hlf * 16) * 128;
#pragma unroll
      for (int j = 0; j < 8; ++j)
        *(bf16x8*)(dst + lane * 8 + j * 256) =
            *(const bf16x8*)(smemO + lane * 8 + j * 256);
    } else {  // kp transposed: kpT[bh][m][l], contiguous along l
#pragma unroll
      for (int n = 0; n < 8; ++n) {
        int m = n * 16 + cn;
        bf16x8 o;
#pragma unroll
        for (int r = 0; r < 8; ++r) o[r] = (bf16_t)(fmaxf(f[n][r], 0.0f) + 1e-3f);
        *(bf16x8*)(kpT + ((size_t)bh * 128 + m) * 4096 + l0 + hlf * 16 + rb) = o;
      }
    }
  }
}

// ---------------------------------------------------------------------------
// ks_sum: row sums of kpT -> row 64 of the extended kvs B-matrix
// ---------------------------------------------------------------------------
__global__ __launch_bounds__(32) void k_kssum(const bf16_t* __restrict__ kpT,
                                              bf16_t* __restrict__ kvs) {
  const int row = blockIdx.x;  // bh*128 + m  (8192 rows)
  const int lane = threadIdx.x;
  const bf16_t* p = kpT + (size_t)row * 4096;
  float s = 0.0f;
  for (int c = lane * 8; c < 4096; c += 32 * 8) {
    bf16x8 v = *(const bf16x8*)(p + c);
#pragma unroll
    for (int i = 0; i < 8; ++i) s += (float)v[i];
  }
#pragma unroll
  for (int off = 16; off > 0; off >>= 1) s += __shfl_xor(s, off, 32);
  if (lane == 0) {
    int bh = row >> 7, m = row & 127;
    kvs[(size_t)bh * 80 * 128 + 64 * 128 + m] = (bf16_t)s;
  }
}

// ---------------------------------------------------------------------------
// kvs partials: kp^T @ v over L/4 chunk. 32 m-rows per wave. grid (4, 64, 4)
// ---------------------------------------------------------------------------
__global__ __launch_bounds__(32) void k_kvs_part(const bf16_t* __restrict__ kpT,
                                                 const bf16_t* __restrict__ vT,
                                                 float* __restrict__ part) {
  const int mt = blockIdx.x, bh = blockIdx.y, sp = blockIdx.z;
  const int lane = threadIdx.x;
  const bf16_t* A = kpT + ((size_t)bh * 128 + mt * 32) * 4096 + sp * 1024;
  const bf16_t* Bp = vT + (size_t)bh * 64 * 4096 + sp * 1024;

  floatx8 acc[8] = {};
  bf16x16 Af[2][2], Bf[4][2];
#define LOAD_KV(p, kk)                                                     \
  do {                                                                     \
    Af[0][p] = ldA(A + (kk), 4096, lane);                                  \
    Af[1][p] = ldA(A + (size_t)16 * 4096 + (kk), 4096, lane);              \
    Bf[0][p] = ldB(Bp + (size_t)0 * 16 * 4096 + (kk), 4096, lane);         \
    Bf[1][p] = ldB(Bp + (size_t)1 * 16 * 4096 + (kk), 4096, lane);         \
    Bf[2][p] = ldB(Bp + (size_t)2 * 16 * 4096 + (kk), 4096, lane);         \
    Bf[3][p] = ldB(Bp + (size_t)3 * 16 * 4096 + (kk), 4096, lane);         \
  } while (0)
#define COMP_KV(p)                                                         \
  do {                                                                     \
    acc[0] = wmma_bf16(Af[0][p], Bf[0][p], acc[0]);                        \
    acc[1] = wmma_bf16(Af[0][p], Bf[1][p], acc[1]);                        \
    acc[2] = wmma_bf16(Af[0][p], Bf[2][p], acc[2]);                        \
    acc[3] = wmma_bf16(Af[0][p], Bf[3][p], acc[3]);                        \
    acc[4] = wmma_bf16(Af[1][p], Bf[0][p], acc[4]);                        \
    acc[5] = wmma_bf16(Af[1][p], Bf[1][p], acc[5]);                        \
    acc[6] = wmma_bf16(Af[1][p], Bf[2][p], acc[6]);                        \
    acc[7] = wmma_bf16(Af[1][p], Bf[3][p], acc[7]);                        \
  } while (0)
  LOAD_KV(0, 0);
  for (int kk = 0; kk < 1024 - 64; kk += 64) {
    LOAD_KV(1, kk + 32);
    COMP_KV(0);
    LOAD_KV(0, kk + 64);
    COMP_KV(1);
  }
  LOAD_KV(1, 1024 - 32);
  COMP_KV(0);
  COMP_KV(1);
#undef LOAD_KV
#undef COMP_KV

  float* dst = part + (((size_t)sp * 64 + bh) * 128 + mt * 32) * 64;
  const int cn = lane & 15, rb = (lane >> 4) << 3;
#pragma unroll
  for (int hlf = 0; hlf < 2; ++hlf)
#pragma unroll
    for (int n = 0; n < 4; ++n)
#pragma unroll
      for (int r = 0; r < 8; ++r)
        dst[(size_t)(hlf * 16 + rb + r) * 64 + n * 16 + cn] = acc[hlf * 4 + n][r];
}

// reduce partials (and transpose m<->d) into kvs rows 0..63 (bf16)
__global__ void k_kvs_reduce(const float* __restrict__ part, bf16_t* __restrict__ kvs) {
  int o = blockIdx.x * 256 + threadIdx.x;   // 64*64*128
  if (o >= 64 * 64 * 128) return;
  int bh = o >> 13, r = o & 8191, d = r >> 7, m = r & 127;
  float s = 0.0f;
#pragma unroll
  for (int sp = 0; sp < 4; ++sp)
    s += part[(((size_t)sp * 64 + bh) * 128 + m) * 64 + d];
  kvs[(size_t)bh * 80 * 128 + (size_t)d * 128 + m] = (bf16_t)s;
}

// ---------------------------------------------------------------------------
// num + den + normalize: qp[16x128] @ kvs_ext^T -> att tile (bf16)
// kvs row 64 = ks_sum -> den. grid (256, 16, 4)
// ---------------------------------------------------------------------------
__global__ __launch_bounds__(32) void k_num_att(const bf16_t* __restrict__ qp,
                                                const bf16_t* __restrict__ kvs,
                                                bf16_t* __restrict__ att) {
  const int lt = blockIdx.x, h = blockIdx.y, b = blockIdx.z;
  const int lane = threadIdx.x;
  const int bh = b * 16 + h;
  const bf16_t* A = qp + ((size_t)bh * 4096 + lt * 16) * 128;
  const bf16_t* Bp = kvs + (size_t)bh * 80 * 128;

  floatx8 acc[5] = {};
  bf16x16 a = ldA(A, 128, lane);
  bf16x16 bb[5];
#pragma unroll
  for (int n = 0; n < 5; ++n) bb[n] = ldB(Bp + n * 16 * 128, 128, lane);
#pragma unroll
  for (int kk = 32; kk < 128; kk += 32) {
    bf16x16 na = ldA(A + kk, 128, lane);
    bf16x16 nb[5];
#pragma unroll
    for (int n = 0; n < 5; ++n) nb[n] = ldB(Bp + n * 16 * 128 + kk, 128, lane);
#pragma unroll
    for (int n = 0; n < 5; ++n) acc[n] = wmma_bf16(a, bb[n], acc[n]);
    a = na;
#pragma unroll
    for (int n = 0; n < 5; ++n) bb[n] = nb[n];
  }
#pragma unroll
  for (int n = 0; n < 5; ++n) acc[n] = wmma_bf16(a, bb[n], acc[n]);

  bf16_t* dst = att + ((size_t)(b * 4096 + lt * 16)) * 1024 + h * 64;
  const int cn = lane & 15, rb = (lane >> 4) << 3;
#pragma unroll
  for (int r = 0; r < 8; ++r) {
    // den for this row lives in lane 0 (rows 0-7) / lane 16 (rows 8-15), col 0
    int di = __builtin_amdgcn_ds_bpermute((lane & 16) << 2,
                                          __builtin_bit_cast(int, acc[4][r]));
    float rcp = 1.0f / __builtin_bit_cast(float, di);
#pragma unroll
    for (int n = 0; n < 4; ++n)
      dst[(size_t)(rb + r) * 1024 + n * 16 + cn] = (bf16_t)(acc[n][r] * rcp);
  }
}

// ---------------------------------------------------------------------------
// output projection: att[16384x1024] @ Wo -> out fp32. 32 rows/wave.
// grid (512, 16)
// ---------------------------------------------------------------------------
__global__ __launch_bounds__(32) void k_out(const bf16_t* __restrict__ att,
                                            const bf16_t* __restrict__ WoT,
                                            float* __restrict__ out) {
  const int t = blockIdx.x, ns = blockIdx.y;
  const int lane = threadIdx.x;
  const bf16_t* A = att + (size_t)t * 32 * 1024;
  const bf16_t* Bp = WoT + (size_t)ns * 64 * 1024;

  floatx8 acc[8] = {};
  bf16x16 Af[2][2], Bf[4][2];
#define LOAD_O(p, kk)                                            \
  do {                                                           \
    Af[0][p] = ldA(A + (kk), 1024, lane);                        \
    Af[1][p] = ldA(A + 16 * 1024 + (kk), 1024, lane);            \
    Bf[0][p] = ldB(Bp + 0 * 16 * 1024 + (kk), 1024, lane);       \
    Bf[1][p] = ldB(Bp + 1 * 16 * 1024 + (kk), 1024, lane);       \
    Bf[2][p] = ldB(Bp + 2 * 16 * 1024 + (kk), 1024, lane);       \
    Bf[3][p] = ldB(Bp + 3 * 16 * 1024 + (kk), 1024, lane);       \
  } while (0)
#define COMP_O(p)                                                \
  do {                                                           \
    acc[0] = wmma_bf16(Af[0][p], Bf[0][p], acc[0]);              \
    acc[1] = wmma_bf16(Af[0][p], Bf[1][p], acc[1]);              \
    acc[2] = wmma_bf16(Af[0][p], Bf[2][p], acc[2]);              \
    acc[3] = wmma_bf16(Af[0][p], Bf[3][p], acc[3]);              \
    acc[4] = wmma_bf16(Af[1][p], Bf[0][p], acc[4]);              \
    acc[5] = wmma_bf16(Af[1][p], Bf[1][p], acc[5]);              \
    acc[6] = wmma_bf16(Af[1][p], Bf[2][p], acc[6]);              \
    acc[7] = wmma_bf16(Af[1][p], Bf[3][p], acc[7]);              \
  } while (0)
  LOAD_O(0, 0);
  for (int kk = 0; kk < 1024 - 64; kk += 64) {
    __builtin_prefetch(A + (lane & 15) * 1024 + kk + 64, 0, 1);
    LOAD_O(1, kk + 32);
    COMP_O(0);
    LOAD_O(0, kk + 64);
    COMP_O(1);
  }
  LOAD_O(1, 1024 - 32);
  COMP_O(0);
  COMP_O(1);
#undef LOAD_O
#undef COMP_O

  float* dst = out + (size_t)t * 32 * 1024 + ns * 64;
  const int cn = lane & 15, rb = (lane >> 4) << 3;
#pragma unroll
  for (int hlf = 0; hlf < 2; ++hlf)
#pragma unroll
    for (int n = 0; n < 4; ++n)
#pragma unroll
      for (int r = 0; r < 8; ++r)
        dst[(size_t)(hlf * 16 + rb + r) * 1024 + n * 16 + cn] = acc[hlf * 4 + n][r];
}

// ---------------------------------------------------------------------------
extern "C" void kernel_launch(void* const* d_in, const int* in_sizes, int n_in,
                              void* d_out, int out_size, void* d_ws, size_t ws_size,
                              hipStream_t stream) {
  (void)in_sizes; (void)n_in; (void)out_size; (void)ws_size;
  const float* Xq  = (const float*)d_in[0];
  const float* Xs  = (const float*)d_in[1];
  const float* Wq  = (const float*)d_in[2];
  const float* Wk  = (const float*)d_in[3];
  const float* Wv  = (const float*)d_in[4];
  const float* Wo  = (const float*)d_in[5];
  const float* prj = (const float*)d_in[6];
  float* out = (float*)d_out;
  char* ws = (char*)d_ws;

  // workspace layout (bytes)
  bf16_t* qp    = (bf16_t*)(ws + 0);                       //  64 MiB  [bh][l][m]
  bf16_t* kpT   = (bf16_t*)(ws + (size_t)67108864);        //  64 MiB  [bh][m][l]
  bf16_t* att   = kpT;                                     //  alias: kpT dead after k_kvs_part
  bf16_t* vT    = (bf16_t*)(ws + (size_t)134217728);       //  32 MiB  [bh][d][l]
  bf16_t* WqT   = (bf16_t*)(ws + (size_t)167772160);       //   2 MiB
  bf16_t* WkT   = (bf16_t*)(ws + (size_t)169869312);       //   2 MiB
  bf16_t* WvT   = (bf16_t*)(ws + (size_t)171966464);       //   2 MiB
  bf16_t* WoT   = (bf16_t*)(ws + (size_t)174063616);       //   2 MiB
  bf16_t* projb = (bf16_t*)(ws + (size_t)176160768);       //  16 KiB
  bf16_t* kvs   = (bf16_t*)(ws + (size_t)176177152);       // 1.25 MiB [bh][80][128]
  float*  part  = (float*)(ws + (size_t)177487872);        //   8 MiB

  k_conv_wT  <<<4096, 256, 0, stream>>>(Wq, WqT);
  k_conv_wT  <<<4096, 256, 0, stream>>>(Wk, WkT);
  k_conv_wT  <<<4096, 256, 0, stream>>>(Wv, WvT);
  k_conv_woT <<<4096, 256, 0, stream>>>(Wo, WoT);
  k_conv_proj<<<32,   256, 0, stream>>>(prj, projb);
  k_zero_bf16<<<(64 * 80 * 128 + 255) / 256, 256, 0, stream>>>(kvs, 64 * 80 * 128);

  k_qkv_feat<<<dim3(512, 16, 3), 32, 0, stream>>>(Xq, Xs, WqT, WkT, WvT, projb,
                                                  qp, kpT, vT);
  k_kssum   <<<8192, 32, 0, stream>>>(kpT, kvs);
  k_kvs_part<<<dim3(4, 64, 4), 32, 0, stream>>>(kpT, vT, part);
  k_kvs_reduce<<<(64 * 64 * 128 + 255) / 256, 256, 0, stream>>>(part, kvs);
  k_num_att <<<dim3(256, 16, 4), 32, 0, stream>>>(qp, kvs, att);
  k_out     <<<dim3(512, 16), 32, 0, stream>>>(att, WoT, out);
}
